// Experts_79285096284331
// MI455X (gfx1250) — matrix operand order repaired and checked by reading, not compiled, not run on hardware
//
#include <hip/hip_runtime.h>
#include <hip/hip_bf16.h>

typedef __attribute__((ext_vector_type(16))) __bf16 v16bf;
typedef __attribute__((ext_vector_type(8)))  __bf16 v8bf;
typedef __attribute__((ext_vector_type(4)))  __bf16 v4bf;
typedef __attribute__((ext_vector_type(2)))  __bf16 v2bf;
typedef __attribute__((ext_vector_type(8)))  float  v8f;
typedef __attribute__((ext_vector_type(4)))  float  v4f;

#define E_EXP 64
#define N_TOK 4096
#define D_DIM 2048
#define H_DIM 1024
#define T_TOK 64

// LDS row strides in halves; multiples of 8 so every 16B fragment load is aligned.
#define SA 40   // A tile row stride (32 K-halves + pad)
#define SB 40   // B tile row stride, stored transposed [n][k]

union Frag { v16bf v; v8bf h[2]; };

__device__ __forceinline__ v8f wmma_bf16(const v16bf& a, const v16bf& b, v8f c) {
    return __builtin_amdgcn_wmma_f32_16x16x32_bf16(false, a, false, b, (short)0, c,
                                                   false, false);
}

__device__ __forceinline__ void split_bf16(float f, __bf16& hi, __bf16& lo) {
    hi = (__bf16)f;
    lo = (__bf16)(f - (float)hi);
}

// Stage a 2(k) x 4(n) fp32 sub-block of B into transposed [n][k] hi/lo bf16 LDS
// planes using packed b32 stores ({k,k+1} bf16 pairs).
__device__ __forceinline__ void stage_b_pair(const v4f& va, const v4f& vb,
                                             __bf16* __restrict__ bhi,
                                             __bf16* __restrict__ blo,
                                             int n0, int k2) {
    #pragma unroll
    for (int i = 0; i < 4; ++i) {
        __bf16 h0, l0, h1, l1;
        split_bf16(va[i], h0, l0);
        split_bf16(vb[i], h1, l1);
        v2bf ph = {h0, h1};
        v2bf pl = {l0, l1};
        *(v2bf*)(bhi + (n0 + i) * SB + 2 * k2) = ph;
        *(v2bf*)(blo + (n0 + i) * SB + 2 * k2) = pl;
    }
}

// ---------------------------------------------------------------------------
// Kernel 1: gate/up GEMMs + SwiGLU.  Grid: (H/64, E).  Block: 256 (8 waves).
// Each WG: 64(M) x 64(N) tile of hidden for one expert; K loop over D=2048.
// ---------------------------------------------------------------------------
__global__ __launch_bounds__(256) void moe_gate_up_kernel(
    const float* __restrict__ x,
    const float* __restrict__ gate_w,
    const float* __restrict__ up_w,
    __bf16* __restrict__ hid_hi,
    __bf16* __restrict__ hid_lo)
{
    const int e    = blockIdx.y;
    const int nblk = blockIdx.x;          // 16 blocks of 64 H-columns
    const int tid  = threadIdx.x;
    const int wave = tid >> 5;
    const int lane = tid & 31;
    const int hlf  = lane >> 4;           // which 16-lane half
    const int l16  = lane & 15;

    __shared__ __align__(16) __bf16 lAhi[T_TOK * SA];
    __shared__ __align__(16) __bf16 lAlo[T_TOK * SA];
    __shared__ __align__(16) __bf16 lGhi[64 * SB];
    __shared__ __align__(16) __bf16 lGlo[64 * SB];
    __shared__ __align__(16) __bf16 lUhi[64 * SB];
    __shared__ __align__(16) __bf16 lUlo[64 * SB];

    const int mt = wave & 3;              // M tile (16 rows) 0..3
    const int nh = wave >> 2;             // n-tile pair 0..1

    v8f accg[2] = {};
    v8f accu[2] = {};

    const float* xe = x + (size_t)(e * T_TOK) * D_DIM;
    const float* gw = gate_w + (size_t)e * D_DIM * H_DIM + nblk * 64;
    const float* uw = up_w   + (size_t)e * D_DIM * H_DIM + nblk * 64;

    // B staging decomposition: 2(k) x 4(n) per thread
    const int k2 = tid >> 4;              // 0..15 -> rows 2k2, 2k2+1
    const int n0 = (tid & 15) << 2;       // 0..60

    for (int k0 = 0; k0 < D_DIM; k0 += 32) {
        __syncthreads();
        // ---- stage A: x[e, 0:64, k0:k0+32] fp32 -> bf16 hi/lo, row-major ----
        #pragma unroll
        for (int j = 0; j < 2; ++j) {
            int idx = tid + 256 * j;              // 0..511 over 64x8 v4f
            int r   = idx >> 3;                   // row 0..63
            int c4  = (idx & 7) << 2;             // 0..28
            v4f v = *(const v4f*)(xe + (size_t)r * D_DIM + k0 + c4);
            v4bf h4, l4;
            #pragma unroll
            for (int i = 0; i < 4; ++i) {
                __bf16 h, l;
                split_bf16(v[i], h, l);
                h4[i] = h;
                l4[i] = l;
            }
            *(v4bf*)(lAhi + r * SA + c4) = h4;
            *(v4bf*)(lAlo + r * SA + c4) = l4;
        }
        // ---- stage B (gate & up): 32k x 64n tile -> transposed [n][k] ----
        {
            const float* gp = gw + (size_t)(k0 + 2 * k2) * H_DIM + n0;
            const float* up = uw + (size_t)(k0 + 2 * k2) * H_DIM + n0;
            v4f g0 = *(const v4f*)(gp);
            v4f g1 = *(const v4f*)(gp + H_DIM);
            v4f u0 = *(const v4f*)(up);
            v4f u1 = *(const v4f*)(up + H_DIM);
            if (k0 + 32 < D_DIM) {                 // prefetch next k-tile
                __builtin_prefetch(gp + 32 * H_DIM, 0, 0);
                __builtin_prefetch(up + 32 * H_DIM, 0, 0);
            }
            stage_b_pair(g0, g1, lGhi, lGlo, n0, k2);
            stage_b_pair(u0, u1, lUhi, lUlo, n0, k2);
        }
        __syncthreads();

        // ---- A fragments (16x32 bf16 layout: lane<16 K{0-7,16-23}) ----
        Frag ahi, alo;
        {
            int base = (mt * 16 + l16) * SA + hlf * 8;
            ahi.h[0] = *(const v8bf*)(lAhi + base);
            ahi.h[1] = *(const v8bf*)(lAhi + base + 16);
            alo.h[0] = *(const v8bf*)(lAlo + base);
            alo.h[1] = *(const v8bf*)(lAlo + base + 16);
        }
        #pragma unroll
        for (int t = 0; t < 2; ++t) {
            int nt = nh * 2 + t;
            // B fragment: lane<16 -> N=l16, K{0-7,16-23}; lane>=16 K{8-15,24-31}
            int nb = (nt * 16 + l16) * SB + hlf * 8;
            Frag bh, bl;
            bh.h[0] = *(const v8bf*)(lGhi + nb);
            bh.h[1] = *(const v8bf*)(lGhi + nb + 16);
            bl.h[0] = *(const v8bf*)(lGlo + nb);
            bl.h[1] = *(const v8bf*)(lGlo + nb + 16);
            accg[t] = wmma_bf16(ahi.v, bh.v, accg[t]);
            accg[t] = wmma_bf16(ahi.v, bl.v, accg[t]);
            accg[t] = wmma_bf16(alo.v, bh.v, accg[t]);
            bh.h[0] = *(const v8bf*)(lUhi + nb);
            bh.h[1] = *(const v8bf*)(lUhi + nb + 16);
            bl.h[0] = *(const v8bf*)(lUlo + nb);
            bl.h[1] = *(const v8bf*)(lUlo + nb + 16);
            accu[t] = wmma_bf16(ahi.v, bh.v, accu[t]);
            accu[t] = wmma_bf16(ahi.v, bl.v, accu[t]);
            accu[t] = wmma_bf16(alo.v, bh.v, accu[t]);
        }
    }

    // ---- SwiGLU in fp32, write hidden as bf16 hi/lo planes ----
    // C/D layout: VGPR r -> (M = r + 8*hlf, N = l16)
    #pragma unroll
    for (int t = 0; t < 2; ++t) {
        int ncol = nblk * 64 + (nh * 2 + t) * 16 + l16;
        #pragma unroll
        for (int r = 0; r < 8; ++r) {
            int   row = mt * 16 + r + hlf * 8;
            float g   = accg[t][r];
            float u   = accu[t][r];
            float s   = __fdividef(g, 1.0f + __expf(-g));   // fast silu
            float hv  = s * u;
            __bf16 hh, hl;
            split_bf16(hv, hh, hl);
            size_t off = ((size_t)(e * T_TOK) + row) * H_DIM + ncol;
            hid_hi[off] = hh;
            hid_lo[off] = hl;
        }
    }
}

// ---------------------------------------------------------------------------
// Kernel 2: down projection.  Grid: (D/64, E).  Block: 256 (8 waves).
// out[e, 0:64, n0:n0+64] = hidden[e] (64x1024) @ down_w[e] (1024x2048)
// ---------------------------------------------------------------------------
__global__ __launch_bounds__(256) void moe_down_kernel(
    const __bf16* __restrict__ hid_hi,
    const __bf16* __restrict__ hid_lo,
    const float* __restrict__ down_w,
    float* __restrict__ out)
{
    const int e    = blockIdx.y;
    const int nblk = blockIdx.x;          // 32 blocks of 64 D-columns
    const int tid  = threadIdx.x;
    const int wave = tid >> 5;
    const int lane = tid & 31;
    const int hlf  = lane >> 4;
    const int l16  = lane & 15;

    __shared__ __align__(16) __bf16 lAhi[T_TOK * SA];
    __shared__ __align__(16) __bf16 lAlo[T_TOK * SA];
    __shared__ __align__(16) __bf16 lBhi[64 * SB];
    __shared__ __align__(16) __bf16 lBlo[64 * SB];

    const int mt = wave & 3;
    const int nh = wave >> 2;

    v8f acc[2] = {};

    const __bf16* ahg = hid_hi + (size_t)(e * T_TOK) * H_DIM;
    const __bf16* alg = hid_lo + (size_t)(e * T_TOK) * H_DIM;
    const float*  dw  = down_w + (size_t)e * H_DIM * D_DIM + nblk * 64;

    const int k2 = tid >> 4;              // 0..15
    const int n0 = (tid & 15) << 2;       // 0..60

    for (int k0 = 0; k0 < H_DIM; k0 += 32) {
        __syncthreads();
        // ---- stage A: hidden bf16 planes, 64x32 halves each ----
        {
            int r  = tid >> 2;                    // row 0..63
            int c8 = (tid & 3) << 3;              // 0,8,16,24
            size_t g = (size_t)r * H_DIM + k0 + c8;
            *(v8bf*)(lAhi + r * SA + c8) = *(const v8bf*)(ahg + g);
            *(v8bf*)(lAlo + r * SA + c8) = *(const v8bf*)(alg + g);
        }
        // ---- stage B: down_w[e, k0:k0+32, n0:n0+64] -> transposed [n][k] ----
        {
            const float* dp = dw + (size_t)(k0 + 2 * k2) * D_DIM + n0;
            v4f d0 = *(const v4f*)(dp);
            v4f d1 = *(const v4f*)(dp + D_DIM);
            if (k0 + 32 < H_DIM) {
                __builtin_prefetch(dp + 32 * D_DIM, 0, 0);
            }
            stage_b_pair(d0, d1, lBhi, lBlo, n0, k2);
        }
        __syncthreads();

        Frag ahi, alo;
        {
            int base = (mt * 16 + l16) * SA + hlf * 8;
            ahi.h[0] = *(const v8bf*)(lAhi + base);
            ahi.h[1] = *(const v8bf*)(lAhi + base + 16);
            alo.h[0] = *(const v8bf*)(lAlo + base);
            alo.h[1] = *(const v8bf*)(lAlo + base + 16);
        }
        #pragma unroll
        for (int t = 0; t < 2; ++t) {
            int nt = nh * 2 + t;
            int nb = (nt * 16 + l16) * SB + hlf * 8;
            Frag bh, bl;
            bh.h[0] = *(const v8bf*)(lBhi + nb);
            bh.h[1] = *(const v8bf*)(lBhi + nb + 16);
            bl.h[0] = *(const v8bf*)(lBlo + nb);
            bl.h[1] = *(const v8bf*)(lBlo + nb + 16);
            acc[t] = wmma_bf16(ahi.v, bh.v, acc[t]);
            acc[t] = wmma_bf16(ahi.v, bl.v, acc[t]);
            acc[t] = wmma_bf16(alo.v, bh.v, acc[t]);
        }
    }

    #pragma unroll
    for (int t = 0; t < 2; ++t) {
        int ncol = nblk * 64 + (nh * 2 + t) * 16 + l16;
        #pragma unroll
        for (int r = 0; r < 8; ++r) {
            int row = e * T_TOK + mt * 16 + r + hlf * 8;
            out[(size_t)row * D_DIM + ncol] = acc[t][r];
        }
    }
}

// ---------------------------------------------------------------------------
extern "C" void kernel_launch(void* const* d_in, const int* in_sizes, int n_in,
                              void* d_out, int out_size, void* d_ws, size_t ws_size,
                              hipStream_t stream) {
    const float* x  = (const float*)d_in[0];
    // d_in[1] = tokens_per_expert: equal split (64 per expert), not needed.
    const float* gw = (const float*)d_in[2];
    const float* uw = (const float*)d_in[3];
    const float* dw = (const float*)d_in[4];
    float* out = (float*)d_out;

    // workspace: hidden as bf16 hi/lo planes, 2 x (4096*1024*2B) = 16 MB
    __bf16* hid_hi = (__bf16*)d_ws;
    __bf16* hid_lo = hid_hi + (size_t)N_TOK * H_DIM;

    dim3 g1(H_DIM / 64, E_EXP);   // 16 x 64
    dim3 g2(D_DIM / 64, E_EXP);   // 32 x 64
    moe_gate_up_kernel<<<g1, 256, 0, stream>>>(x, gw, uw, hid_hi, hid_lo);
    moe_down_kernel<<<g2, 256, 0, stream>>>(hid_hi, hid_lo, dw, out);
}
